// TransformerBlock_2396591751715
// MI455X (gfx1250) — matrix operand order repaired and checked by reading, not compiled, not run on hardware
//
#include <hip/hip_runtime.h>

// ---------------------------------------------------------------------------
// Types for CDNA5 WMMA (wave32): v16bf A/B fragments, v8f C/D accumulator
// ---------------------------------------------------------------------------
typedef __attribute__((ext_vector_type(16))) __bf16 bf16x16;
typedef __attribute__((ext_vector_type(8)))  float  v8f;

union FragAB { bf16x16 v; uint4 u[2]; };

__device__ __forceinline__ v8f wmma_bf16(const bf16x16 a, const bf16x16 b, const v8f c) {
    // D = A(16x32 bf16) * B(32x16 bf16) + C(16x16 f32)
    return __builtin_amdgcn_wmma_f32_16x16x32_bf16(
        /*neg_a=*/false, a, /*neg_b=*/false, b,
        /*c_mod=*/(short)0, c, /*reuse_a=*/false, /*reuse_b=*/false);
}

__device__ __forceinline__ v8f zero8() {
    v8f z;
#pragma unroll
    for (int i = 0; i < 8; ++i) z[i] = 0.0f;
    return z;
}

// fp32 -> bf16 round-to-nearest-even via bit manipulation
__device__ __forceinline__ unsigned short f2bf_bits(float f) {
    unsigned int u = __float_as_uint(f);
    u += 0x7FFFu + ((u >> 16) & 1u);
    return (unsigned short)(u >> 16);
}
__device__ __forceinline__ __bf16 f2bf(float f) {
    union { unsigned short u; __bf16 b; } x;
    x.u = f2bf_bits(f);
    return x.b;
}

// LDS byte offset of a shared-memory pointer (flat->LDS uses addr[31:0])
__device__ __forceinline__ unsigned lds_off(const void* p) {
    return (unsigned)(unsigned long long)p;
}

// Per-lane async DMA global->LDS, 16 bytes. Tracked by ASYNCcnt.
// the instruction offset is added to both the LDS and the global address
__device__ __forceinline__ void async_ld_b128(unsigned ldsaddr, const void* gaddr) {
    asm volatile("global_load_async_to_lds_b128 %0, %1, off"
                 :: "v"(ldsaddr), "v"(gaddr) : "memory");
}
__device__ __forceinline__ void async_ld_b128_off16(unsigned ldsaddr, const void* gaddr) {
    asm volatile("global_load_async_to_lds_b128 %0, %1, off offset:16"
                 :: "v"(ldsaddr), "v"(gaddr) : "memory");
}
__device__ __forceinline__ void wait_async0() {
    asm volatile("s_wait_asynccnt 0x0" ::: "memory");
}

// 16-lane row reductions (C/D layout: cols 0..15 live in lanes 0..15 and
// 16..31 independently; xor masks < 16 stay within each half).
__device__ __forceinline__ float rowmax16(float x) {
    x = fmaxf(x, __shfl_xor(x, 1));
    x = fmaxf(x, __shfl_xor(x, 2));
    x = fmaxf(x, __shfl_xor(x, 4));
    x = fmaxf(x, __shfl_xor(x, 8));
    return x;
}
__device__ __forceinline__ float rowsum16(float x) {
    x += __shfl_xor(x, 1);
    x += __shfl_xor(x, 2);
    x += __shfl_xor(x, 4);
    x += __shfl_xor(x, 8);
    return x;
}

// ---------------------------------------------------------------------------
// fp32 -> bf16 conversion kernel
// ---------------------------------------------------------------------------
__global__ __launch_bounds__(256) void cvt_f32_bf16(const float* __restrict__ in,
                                                    __bf16* __restrict__ out, int n) {
    int i = blockIdx.x * 256 + threadIdx.x;
    if (i < n) out[i] = f2bf(in[i]);
}

// ---------------------------------------------------------------------------
// Tiled bf16 GEMM: C[M,N] = A[M,K] @ B[K,N] (+bias, opt ReLU, opt scale)
// Block tile 128x64, K-step 32. 256 threads = 8 waves, each wave 32x32
// (2x2 WMMA tiles). Double-buffered LDS; A tile staged by async-to-LDS DMA,
// B tile (transposed) prefetched into registers one step ahead.
// One barrier per K-step: per-wave program order store(t)->bar(t)->compute(t)
// guarantees no buffer is overwritten while any wave still reads it.
// flags: 1=ReLU, 2=store fp32, 4=store bf16
// ---------------------------------------------------------------------------
__global__ __launch_bounds__(256)
void gemm_bf16_kernel(const __bf16* __restrict__ A, const __bf16* __restrict__ B,
                      const float* __restrict__ bias,
                      float* __restrict__ outF, __bf16* __restrict__ outB,
                      int M, int N, int K, float scale, int flags) {
    __shared__ __bf16 As[2][128 * 40];
    __shared__ __bf16 Bs[2][64 * 40];

    const int tid  = threadIdx.x;
    const int lane = tid & 31;
    const int w    = tid >> 5;
    const int waveM = w & 3;   // 0..3 -> rows waveM*32
    const int waveN = w >> 2;  // 0..1 -> cols waveN*32
    const int r15  = lane & 15;
    const int half = lane >> 4;      // 0 or 1
    const int kb8  = half * 8;       // A-frag K base (per 16-wide half)
    const int kb16 = half * 16;      // unused (B frags also use kb8)

    const int m0 = blockIdx.x * 128;
    const int n0 = blockIdx.y * 64;

    // staging indices
    const int aRow = tid >> 1;            // 0..127
    const int aCol = (tid & 1) * 16;      // 0 or 16
    const int bK   = tid >> 3;            // 0..31
    const int bN   = (tid & 7) * 8;       // 0..56

    const unsigned ldsA0 = lds_off(&As[0][aRow * 40 + aCol]);
    const unsigned ldsA1 = lds_off(&As[1][aRow * 40 + aCol]);
    const __bf16* gA = A + (size_t)(m0 + aRow) * K + aCol;
    const __bf16* gB = B + (size_t)bK * N + n0 + bN;

    union { uint4 u; __bf16 h[8]; } breg;

    // prologue: async A(0), prefetch B(0)
    async_ld_b128(ldsA0, gA);
    async_ld_b128_off16(ldsA0, gA);
    breg.u = *reinterpret_cast<const uint4*>(gB);

    v8f acc[2][2];
#pragma unroll
    for (int mt = 0; mt < 2; ++mt)
#pragma unroll
        for (int nt = 0; nt < 2; ++nt) acc[mt][nt] = zero8();

    const int T = K >> 5;
    for (int t = 0; t < T; ++t) {
        const int buf = t & 1;
        // commit B(t) transpose to LDS: Bs[n][k] = B[k][n]
#pragma unroll
        for (int j = 0; j < 8; ++j) Bs[buf][(bN + j) * 40 + bK] = breg.h[j];
        wait_async0();       // A(t) DMA complete (this wave's portion)
        __syncthreads();     // all waves' A(t)+B(t) staging visible

        if (t + 1 < T) {     // launch next tile's staging under this tile's math
            const int k1 = (t + 1) << 5;
            const unsigned ldsA = (buf ? ldsA0 : ldsA1);
            async_ld_b128(ldsA, gA + k1);
            async_ld_b128_off16(ldsA, gA + k1);
            breg.u = *reinterpret_cast<const uint4*>(gB + (size_t)k1 * N);
        }

        FragAB af[2], bf[2];
#pragma unroll
        for (int mt = 0; mt < 2; ++mt) {
            const int row = waveM * 32 + mt * 16 + r15;
            af[mt].u[0] = *reinterpret_cast<const uint4*>(&As[buf][row * 40 + kb8]);
            af[mt].u[1] = *reinterpret_cast<const uint4*>(&As[buf][row * 40 + kb8 + 16]);
        }
#pragma unroll
        for (int nt = 0; nt < 2; ++nt) {
            const int col = waveN * 32 + nt * 16 + r15;
            bf[nt].u[0] = *reinterpret_cast<const uint4*>(&Bs[buf][col * 40 + kb8]);
            bf[nt].u[1] = *reinterpret_cast<const uint4*>(&Bs[buf][col * 40 + kb8 + 16]);
        }
#pragma unroll
        for (int mt = 0; mt < 2; ++mt)
#pragma unroll
            for (int nt = 0; nt < 2; ++nt)
                acc[mt][nt] = wmma_bf16(af[mt].v, bf[nt].v, acc[mt][nt]);
    }

    // epilogue
#pragma unroll
    for (int nt = 0; nt < 2; ++nt) {
        const int gc = n0 + waveN * 32 + nt * 16 + r15;
        const float bv = bias[gc];
#pragma unroll
        for (int mt = 0; mt < 2; ++mt) {
#pragma unroll
            for (int i = 0; i < 8; ++i) {
                const int gr = m0 + waveM * 32 + mt * 16 + i + half * 8;
                float v = (acc[mt][nt][i] + bv) * scale;
                if (flags & 1) v = fmaxf(v, 0.0f);
                const size_t idx = (size_t)gr * N + gc;
                if (flags & 2) outF[idx] = v;
                if (flags & 4) outB[idx] = f2bf(v);
            }
        }
    }
}

// ---------------------------------------------------------------------------
// Flash attention (bf16 Q/K/V, fp32 online softmax, bf16 ctx out).
// Q already includes bias and the 1/sqrt(dk) scale.
// Block: 256 threads = 8 waves; each wave owns 16 query rows of one (b,h).
// K tile staged by async-to-LDS DMA (already in the B-fragment layout);
// V tile transposed through VGPRs. grid = (S/128, B*H)
// ---------------------------------------------------------------------------
__global__ __launch_bounds__(256)
void flash_attn_kernel(const __bf16* __restrict__ Q, const __bf16* __restrict__ Kmat,
                       const __bf16* __restrict__ V, __bf16* __restrict__ ctx) {
    const int S = 2048, D = 1024;
    __shared__ __bf16 Ks[32 * 72];       // Ks[key][ch]  (exactly the QK^T B layout)
    __shared__ __bf16 Vs[64 * 40];       // Vs[ch][key]  (transposed, P@V B layout)
    __shared__ __bf16 Ps[8 * 16 * 40];   // per-wave P tile, row major [16][32+pad]

    const int tid  = threadIdx.x;
    const int lane = tid & 31;
    const int w    = tid >> 5;
    const int r15  = lane & 15;
    const int half = lane >> 4;
    const int kb8  = half * 8;
    const int kb16 = half * 16;

    const int bh   = blockIdx.y;
    const int b    = bh >> 4;
    const int h    = bh & 15;
    const size_t rowBase = (size_t)b * S;
    const int hcol = h * 64;
    const int q0   = blockIdx.x * 128 + w * 16;

    // staging indices
    const int sk = tid >> 3;        // key row 0..31
    const int sc = (tid & 7) * 8;   // channel 0..56
    const unsigned ldsK = lds_off(&Ks[sk * 72 + sc]);

    // Q fragments (16 rows x 64 ch = 2 K-steps), loaded once
    FragAB qf[2];
    {
        const __bf16* qrow = Q + (rowBase + q0 + r15) * D + hcol;
#pragma unroll
        for (int t = 0; t < 2; ++t) {
            qf[t].u[0] = *reinterpret_cast<const uint4*>(qrow + t * 32 + kb8);
            qf[t].u[1] = *reinterpret_cast<const uint4*>(qrow + t * 32 + kb8 + 16);
        }
    }

    float m[8], l[8];
#pragma unroll
    for (int i = 0; i < 8; ++i) { m[i] = -INFINITY; l[i] = 0.0f; }
    v8f o[4];
#pragma unroll
    for (int nt = 0; nt < 4; ++nt) o[nt] = zero8();

    for (int s0 = 0; s0 < S; s0 += 32) {
        // stage K tile via async DMA: Ks[key][ch]
        async_ld_b128(ldsK, Kmat + (rowBase + s0 + sk) * D + hcol + sc);
        // stage V tile transposed through VGPRs: Vs[ch][key]
        union { uint4 u; __bf16 hh[8]; } vv;
        vv.u = *reinterpret_cast<const uint4*>(V + (rowBase + s0 + sk) * D + hcol + sc);
#pragma unroll
        for (int j = 0; j < 8; ++j) Vs[(sc + j) * 40 + sk] = vv.hh[j];
        wait_async0();
        __syncthreads();

        // scores: two 16x16 tiles (keys s0..s0+15 and s0+16..s0+31)
        v8f sc0 = zero8(), sc1 = zero8();
#pragma unroll
        for (int t = 0; t < 2; ++t) {
            FragAB kf0, kf1;
            kf0.u[0] = *reinterpret_cast<const uint4*>(&Ks[r15 * 72 + t * 32 + kb8]);
            kf0.u[1] = *reinterpret_cast<const uint4*>(&Ks[r15 * 72 + t * 32 + kb8 + 16]);
            kf1.u[0] = *reinterpret_cast<const uint4*>(&Ks[(16 + r15) * 72 + t * 32 + kb8]);
            kf1.u[1] = *reinterpret_cast<const uint4*>(&Ks[(16 + r15) * 72 + t * 32 + kb8 + 16]);
            sc0 = wmma_bf16(qf[t].v, kf0.v, sc0);
            sc1 = wmma_bf16(qf[t].v, kf1.v, sc1);
        }

        // online softmax over the 32 keys of this step
#pragma unroll
        for (int i = 0; i < 8; ++i) {
            float mt = rowmax16(fmaxf(sc0[i], sc1[i]));
            float mn = fmaxf(m[i], mt);
            float alpha = __expf(m[i] - mn);
            float e0 = __expf(sc0[i] - mn);
            float e1 = __expf(sc1[i] - mn);
            float rs = rowsum16(e0 + e1);
            l[i] = l[i] * alpha + rs;
            m[i] = mn;
#pragma unroll
            for (int nt = 0; nt < 4; ++nt) o[nt][i] *= alpha;
            const int prow = i + half * 8;
            Ps[w * 640 + prow * 40 + r15]      = f2bf(e0);
            Ps[w * 640 + prow * 40 + 16 + r15] = f2bf(e1);
        }
        // wave-local LDS turnaround: make P stores visible before A-frag loads
        asm volatile("s_wait_dscnt 0x0" ::: "memory");

        FragAB pf;
        pf.u[0] = *reinterpret_cast<const uint4*>(&Ps[w * 640 + r15 * 40 + kb8]);
        pf.u[1] = *reinterpret_cast<const uint4*>(&Ps[w * 640 + r15 * 40 + kb8 + 16]);
#pragma unroll
        for (int nt = 0; nt < 4; ++nt) {
            FragAB vf;
            vf.u[0] = *reinterpret_cast<const uint4*>(&Vs[(nt * 16 + r15) * 40 + kb8]);
            vf.u[1] = *reinterpret_cast<const uint4*>(&Vs[(nt * 16 + r15) * 40 + kb8 + 16]);
            o[nt] = wmma_bf16(pf.v, vf.v, o[nt]);
        }
        __syncthreads();
    }

    // normalize and write ctx (bf16)
#pragma unroll
    for (int i = 0; i < 8; ++i) {
        const float inv = 1.0f / l[i];
        const size_t grow = rowBase + q0 + i + half * 8;
#pragma unroll
        for (int nt = 0; nt < 4; ++nt)
            ctx[grow * D + hcol + nt * 16 + r15] = f2bf(o[nt][i] * inv);
    }
}

// ---------------------------------------------------------------------------
// Residual add + LayerNorm over rows of 1024. One block (256 thr) per row.
// ---------------------------------------------------------------------------
__global__ __launch_bounds__(256)
void add_ln_kernel(const float* __restrict__ a, const float* __restrict__ r,
                   const float* __restrict__ g, const float* __restrict__ be,
                   float* __restrict__ outF, __bf16* __restrict__ outB) {
    const int row = blockIdx.x;
    const int tid = threadIdx.x;
    const int idx = tid * 4;
    const size_t base = (size_t)row * 1024;

    const float4 va = *reinterpret_cast<const float4*>(a + base + idx);
    const float4 vr = *reinterpret_cast<const float4*>(r + base + idx);
    float v0 = va.x + vr.x, v1 = va.y + vr.y, v2 = va.z + vr.z, v3 = va.w + vr.w;

    float s  = v0 + v1 + v2 + v3;
    float sq = v0 * v0 + v1 * v1 + v2 * v2 + v3 * v3;
#pragma unroll
    for (int msk = 1; msk < 32; msk <<= 1) {
        s  += __shfl_xor(s, msk);
        sq += __shfl_xor(sq, msk);
    }
    __shared__ float sh[16];
    const int w = tid >> 5, lane = tid & 31;
    if (lane == 0) { sh[w] = s; sh[8 + w] = sq; }
    __syncthreads();
    if (tid == 0) {
        float ts = 0.0f, tq = 0.0f;
#pragma unroll
        for (int j = 0; j < 8; ++j) { ts += sh[j]; tq += sh[8 + j]; }
        sh[0] = ts; sh[8] = tq;
    }
    __syncthreads();
    const float mu   = sh[0] * (1.0f / 1024.0f);
    const float var  = sh[8] * (1.0f / 1024.0f) - mu * mu;
    const float rstd = rsqrtf(var + 1e-5f);

    const float4 gg = *reinterpret_cast<const float4*>(g + idx);
    const float4 bb = *reinterpret_cast<const float4*>(be + idx);
    float y0 = (v0 - mu) * rstd * gg.x + bb.x;
    float y1 = (v1 - mu) * rstd * gg.y + bb.y;
    float y2 = (v2 - mu) * rstd * gg.z + bb.z;
    float y3 = (v3 - mu) * rstd * gg.w + bb.w;

    if (outF) {
        float4 o; o.x = y0; o.y = y1; o.z = y2; o.w = y3;
        *reinterpret_cast<float4*>(outF + base + idx) = o;
    }
    if (outB) {
        outB[base + idx + 0] = f2bf(y0);
        outB[base + idx + 1] = f2bf(y1);
        outB[base + idx + 2] = f2bf(y2);
        outB[base + idx + 3] = f2bf(y3);
    }
}

// ---------------------------------------------------------------------------
// Launcher
// ---------------------------------------------------------------------------
extern "C" void kernel_launch(void* const* d_in, const int* in_sizes, int n_in,
                              void* d_out, int out_size, void* d_ws, size_t ws_size,
                              hipStream_t stream) {
    (void)in_sizes; (void)n_in; (void)out_size; (void)ws_size;
    const float* x   = (const float*)d_in[0];
    const float* Wq  = (const float*)d_in[1];
    const float* bq  = (const float*)d_in[2];
    const float* Wk  = (const float*)d_in[3];
    const float* bk  = (const float*)d_in[4];
    const float* Wv  = (const float*)d_in[5];
    const float* bv  = (const float*)d_in[6];
    const float* Wo  = (const float*)d_in[7];
    const float* bo  = (const float*)d_in[8];
    const float* W1  = (const float*)d_in[9];
    const float* b1  = (const float*)d_in[10];
    const float* W2  = (const float*)d_in[11];
    const float* b2  = (const float*)d_in[12];
    const float* g1  = (const float*)d_in[13];
    const float* be1 = (const float*)d_in[14];
    const float* g2  = (const float*)d_in[15];
    const float* be2 = (const float*)d_in[16];

    const int M = 8192;            // B*S
    const int D = 1024, FF = 4096;

    char* ws = (char*)d_ws;
    // Liveness-overlapped layout (bytes):
    //   [0, 64M) : xb | Qb | Kb | Vb  -> later reused entirely for FF1 (64 MB)
    __bf16* xb   = (__bf16*)(ws + 0);
    __bf16* Qb   = (__bf16*)(ws + 16777216);
    __bf16* Kb   = (__bf16*)(ws + 33554432);
    __bf16* Vb   = (__bf16*)(ws + 50331648);
    __bf16* ff1b = (__bf16*)(ws + 0);           // reuses xb/Q/K/V (dead by then)
    __bf16* wqb  = (__bf16*)(ws + 67108864);
    __bf16* wkb  = (__bf16*)(ws + 69206016);
    __bf16* wvb  = (__bf16*)(ws + 71303168);
    __bf16* wob  = (__bf16*)(ws + 73400320);
    __bf16* w1b  = (__bf16*)(ws + 75497472);
    __bf16* w2b  = (__bf16*)(ws + 83886080);
    __bf16* ctx  = (__bf16*)(ws + 92274688);
    float*  gout = (float*) (ws + 109051904);   // attn_out, later ff2_out
    float*  hbuf = (float*) (ws + 142606336);
    __bf16* hb   = (__bf16*)(ws + 176160768);

    // 1) fp32 -> bf16 conversions
    cvt_f32_bf16<<<(8388608 + 255) / 256, 256, 0, stream>>>(x,  xb,  8388608);
    cvt_f32_bf16<<<(1048576 + 255) / 256, 256, 0, stream>>>(Wq, wqb, 1048576);
    cvt_f32_bf16<<<(1048576 + 255) / 256, 256, 0, stream>>>(Wk, wkb, 1048576);
    cvt_f32_bf16<<<(1048576 + 255) / 256, 256, 0, stream>>>(Wv, wvb, 1048576);
    cvt_f32_bf16<<<(1048576 + 255) / 256, 256, 0, stream>>>(Wo, wob, 1048576);
    cvt_f32_bf16<<<(4194304 + 255) / 256, 256, 0, stream>>>(W1, w1b, 4194304);
    cvt_f32_bf16<<<(4194304 + 255) / 256, 256, 0, stream>>>(W2, w2b, 4194304);

    // 2) Q/K/V projections (Q folds the 1/sqrt(dk)=0.125 scale)
    dim3 gP(M / 128, D / 64);
    gemm_bf16_kernel<<<gP, 256, 0, stream>>>(xb, wqb, bq, nullptr, Qb, M, D, D, 0.125f, 4);
    gemm_bf16_kernel<<<gP, 256, 0, stream>>>(xb, wkb, bk, nullptr, Kb, M, D, D, 1.0f,   4);
    gemm_bf16_kernel<<<gP, 256, 0, stream>>>(xb, wvb, bv, nullptr, Vb, M, D, D, 1.0f,   4);

    // 3) flash attention -> ctx (bf16)
    flash_attn_kernel<<<dim3(2048 / 128, 4 * 16), 256, 0, stream>>>(Qb, Kb, Vb, ctx);

    // 4) output projection (fp32 out)
    gemm_bf16_kernel<<<gP, 256, 0, stream>>>(ctx, wob, bo, gout, nullptr, M, D, D, 1.0f, 2);

    // 5) h = LN(x + attn_out), fp32 + bf16
    add_ln_kernel<<<M, 256, 0, stream>>>(x, gout, g1, be1, hbuf, hb);

    // 6) FF1 = relu(h @ W1 + b1), bf16
    dim3 gF1(M / 128, FF / 64);
    gemm_bf16_kernel<<<gF1, 256, 0, stream>>>(hb, w1b, b1, nullptr, ff1b, M, FF, D, 1.0f, 4 | 1);

    // 7) FF2 = ff1 @ W2 + b2, fp32
    gemm_bf16_kernel<<<gP, 256, 0, stream>>>(ff1b, w2b, b2, gout, nullptr, M, D, FF, 1.0f, 2);

    // 8) out = LN(h + ff2)
    add_ln_kernel<<<M, 256, 0, stream>>>(hbuf, gout, g2, be2, (float*)d_out, nullptr);
}
